// CDRsAttention_1262720385153
// MI455X (gfx1250) — matrix operand order repaired and checked by reading, not compiled, not run on hardware
//
#include <hip/hip_runtime.h>
#include <hip/hip_bf16.h>

// ---------------------------------------------------------------------------
// CDRs attention for MI455X (gfx1250): wave32, v_wmma_f32_16x16x32_bf16,
// double-buffered Tensor Data Mover staging (DMA overlapped with WMMA).
// ---------------------------------------------------------------------------

typedef __attribute__((ext_vector_type(16))) __bf16 v16bf;
typedef __attribute__((ext_vector_type(8)))  float  v8f;
typedef __attribute__((ext_vector_type(4)))  unsigned int v4u;
typedef __attribute__((ext_vector_type(8)))  int v8i;
typedef __attribute__((ext_vector_type(4)))  int v4i;

struct U4 { unsigned int x, y, z, w; };          // 16B plain POD load unit
struct BF16Frag { U4 lo, hi; };                  // 32B = 16 bf16 (one WMMA A/B frag)
struct US16 { unsigned short u[16]; };           // elementwise-assembled frag

static constexpr int BB = 4, TT = 2048, CC = 512, HH = 8, DD = 64;
static constexpr int MM = BB * TT;               // 8192 rows for the big GEMMs
static constexpr float NEGBIG = -1e30f;
static constexpr float L2E = 1.4426950408889634f;

__device__ __forceinline__ unsigned short f2bf(float f) {
  unsigned int u = __builtin_bit_cast(unsigned int, f);
  u += 0x7fffu + ((u >> 16) & 1u);               // round-to-nearest-even
  return (unsigned short)(u >> 16);
}
__device__ __forceinline__ v16bf fragcast(const BF16Frag& f) {
  return __builtin_bit_cast(v16bf, f);
}
__device__ __forceinline__ v16bf fragcast16(const US16& f) {
  return __builtin_bit_cast(v16bf, f);
}
__device__ __forceinline__ v8f vzero() {
  v8f z = {0.f, 0.f, 0.f, 0.f, 0.f, 0.f, 0.f, 0.f};
  return z;
}
__device__ __forceinline__ v8f wmma_bf16(v16bf a, v16bf b, v8f c) {
  return __builtin_amdgcn_wmma_f32_16x16x32_bf16(false, a, false, b, (short)0, c,
                                                 false, false);
}

// Low 32 bits of a generic pointer to LDS == byte offset within LDS
// (ISA 10.2: LDS aperture maps addr[31:0] to LDS).
__device__ __forceinline__ unsigned lds_off(const void* p) {
  return (unsigned)(unsigned long long)p;
}

// ---------------------------------------------------------------------------
// TDM: DMA a 2D tile (tile_w x tile_h elements of 2B, row stride stride_w
// elements) from global memory into contiguous LDS. D# layout per
// CDNA5 ISA. Issue from ONE wave; tracked by TENSORcnt.
// ---------------------------------------------------------------------------
__device__ __forceinline__ void tdm_load_2d_bf16(
    unsigned lds_addr, const void* gaddr,
    unsigned tile_w, unsigned tile_h,
    unsigned tensor_w, unsigned tensor_h, unsigned long long stride_w) {
  unsigned long long ga = (unsigned long long)gaddr;
  v4u g0;
  g0[0] = 1u;                                         // count=1 (valid), user mode
  g0[1] = lds_addr;                                   // lds_addr [63:32]
  g0[2] = (unsigned)ga;                               // global_addr [95:64]
  g0[3] = (unsigned)((ga >> 32) & 0x01FFFFFFu)        // global_addr [120:96]
          | (2u << 30);                               // type=2 ("image")
  v8i g1;
  g1[0] = (int)(1u << 16);                            // data_size=1 (2 bytes)
  g1[1] = (int)((tensor_w & 0xFFFFu) << 16);          // tensor_dim0[15:0]
  g1[2] = (int)((tensor_w >> 16) & 0xFFFFu)           // tensor_dim0[31:16]
          | (int)((tensor_h & 0xFFFFu) << 16);        // tensor_dim1[15:0]
  g1[3] = (int)((tensor_h >> 16) & 0xFFFFu)           // tensor_dim1[31:16]
          | (int)((tile_w & 0xFFFFu) << 16);          // tile_dim0
  g1[4] = (int)(tile_h & 0xFFFFu);                    // tile_dim1; tile_dim2=0
  g1[5] = (int)(unsigned)(stride_w & 0xFFFFFFFFull);  // tensor_dim0_stride[31:0]
  g1[6] = (int)(unsigned)((stride_w >> 32) & 0xFFFFull); // stride[47:32]
  g1[7] = 0;
  v4i g2 = {0, 0, 0, 0}, g3 = {0, 0, 0, 0};
#if __clang_major__ >= 23
  v8i g4 = {0, 0, 0, 0, 0, 0, 0, 0};
  __builtin_amdgcn_tensor_load_to_lds(g0, g1, g2, g3, g4, 0);
#else
  __builtin_amdgcn_tensor_load_to_lds(g0, g1, g2, g3, 0);
#endif
}

// ---------------------------------------------------------------------------
// Prep: f32 [K][N] weight -> bf16 N-major Wt[n][k]  (contiguous along K)
// ---------------------------------------------------------------------------
__global__ __launch_bounds__(256) void transpose_cvt(const float* __restrict__ W,
                                                     unsigned short* __restrict__ Wt) {
  __shared__ float tile[16][17];
  int k0 = blockIdx.y * 16, n0 = blockIdx.x * 16;
  tile[threadIdx.y][threadIdx.x] = W[(size_t)(k0 + threadIdx.y) * CC + n0 + threadIdx.x];
  __syncthreads();
  Wt[(size_t)(n0 + threadIdx.y) * CC + k0 + threadIdx.x] = f2bf(tile[threadIdx.x][threadIdx.y]);
}

// f32 -> bf16 elementwise (x activation)
__global__ __launch_bounds__(256) void cvt_bf16(const float* __restrict__ in,
                                                unsigned short* __restrict__ out, int n) {
  int i = blockIdx.x * blockDim.x + threadIdx.x;
  if (i < n) out[i] = f2bf(in[i]);
}

// Key-bias per (batch, head-class): class 0 = CDR heads (h < 2), class 1 = rest
__global__ __launch_bounds__(256) void prep_bias(const int* __restrict__ mask,
                                                 const int* __restrict__ cdrs,
                                                 float* __restrict__ bias01) {
  int b = blockIdx.x;
  __shared__ int s_any;
  if (threadIdx.x == 0) s_any = 0;
  __syncthreads();
  int local = 0;
  for (int t = threadIdx.x; t < TT; t += blockDim.x)
    if (cdrs[b * TT + t] != 0) local = 1;
  if (local) atomicOr(&s_any, 1);
  __syncthreads();
  int any = s_any;  // sample has at least one CDR token -> CDR mask active
  for (int t = threadIdx.x; t < TT; t += blockDim.x) {
    bool pad = (mask[b * TT + t] == 0);
    bool cdr = any && (cdrs[b * TT + t] == 0);
    bias01[((size_t)b * 2 + 0) * TT + t] = (pad || cdr) ? NEGBIG : 0.0f;
    bias01[((size_t)b * 2 + 1) * TT + t] = pad ? NEGBIG : 0.0f;
  }
}

// ---------------------------------------------------------------------------
// Tiled WMMA GEMM:  out[m][n] = (sum_k A[m][k]*Wt[n][k] + bias[n]) * scale
// Block tile 128(M) x 64(N), 8 waves (one 16-row strip each), K chunks of 32.
// Ping-pong TDM staging: DMA of tile i+1 overlaps WMMA on tile i.
// mode 0: store bf16 scattered to [B][H][T][D] (QKV);  mode 1: f32 [M][N].
// ---------------------------------------------------------------------------
__global__ __launch_bounds__(256) void gemm_wmma_bf16(
    const unsigned short* __restrict__ A,    // [MM][CC] bf16
    const unsigned short* __restrict__ Bt,   // [CC][CC] bf16, N-major
    const float* __restrict__ bias,          // [CC] f32
    unsigned short* __restrict__ outBF, float* __restrict__ outF,
    float scale, int mode) {
  __shared__ alignas(16) unsigned short sA[2][128][32];   // 16 KB
  __shared__ alignas(16) unsigned short sB[2][64][32];    //  8 KB

  const int tid  = threadIdx.x;
  const int wave = tid >> 5, lane = tid & 31;
  const int l16  = lane & 15, hi = lane >> 4;          // hi in {0,1}
  const int m0   = blockIdx.y * 128, n0 = blockIdx.x * 64;
  const int mw   = wave * 16;
  constexpr int KTILES = CC / 32;

  auto issueTile = [&](int kb, int buf) {
    const int k0 = kb * 32;
    tdm_load_2d_bf16(lds_off(&sA[buf][0][0]), A + (size_t)m0 * CC + k0,
                     32, 128, CC, MM, CC);
    tdm_load_2d_bf16(lds_off(&sB[buf][0][0]), Bt + (size_t)n0 * CC + k0,
                     32, 64, CC, CC, CC);
  };

  v8f acc[4];
#pragma unroll
  for (int j = 0; j < 4; ++j) acc[j] = vzero();

  if (wave == 0) issueTile(0, 0);

  for (int kb = 0; kb < KTILES; ++kb) {
    const int cur = kb & 1;
    if (wave == 0) {
      if (kb + 1 < KTILES) {
        issueTile(kb + 1, cur ^ 1);
        // In-order TENSORcnt: <=2 outstanding means tile kb is complete,
        // while tile kb+1's DMA overlaps the WMMAs below.
        __builtin_amdgcn_s_wait_tensorcnt(2);
      } else {
        __builtin_amdgcn_s_wait_tensorcnt(0);
      }
    }
    __syncthreads();

    // A fragment for this wave's 16-row strip
    BF16Frag af;
    af.lo = *(const U4*)&sA[cur][mw + l16][hi * 8];
    af.hi = *(const U4*)&sA[cur][mw + l16][16 + hi * 8];
    v16bf a = fragcast(af);

#pragma unroll
    for (int jn = 0; jn < 4; ++jn) {
      BF16Frag bf_;
      bf_.lo = *(const U4*)&sB[cur][jn * 16 + l16][hi * 8];
      bf_.hi = *(const U4*)&sB[cur][jn * 16 + l16][16 + hi * 8];
      acc[jn] = wmma_bf16(a, fragcast(bf_), acc[jn]);
    }
    __syncthreads();   // WAR: tile kb+2 reuses this buffer
  }

  // epilogue: C/D layout -> lane holds n = l16, rows m = r + 8*hi
#pragma unroll
  for (int jn = 0; jn < 4; ++jn) {
    int gn = n0 + jn * 16 + l16;
    float bn = bias[gn];
#pragma unroll
    for (int r = 0; r < 8; ++r) {
      int gm = m0 + mw + r + hi * 8;
      float val = (acc[jn][r] + bn) * scale;
      if (mode == 0) {
        int b = gm >> 11, t = gm & (TT - 1);
        int h = gn >> 6,  d = gn & (DD - 1);
        outBF[(((size_t)b * HH + h) * TT + t) * DD + d] = f2bf(val);
      } else {
        outF[(size_t)gm * CC + gn] = val;
      }
    }
  }
}

// ---------------------------------------------------------------------------
// Flash attention. Grid: (T/128, B*H). 8 waves; wave owns a 16-query strip.
// Q pre-scaled by 1/sqrt(D). Online softmax with additive key bias.
// Ping-pong TDM staging of K/V tiles; P relayout via per-wave LDS.
// ---------------------------------------------------------------------------
__global__ __launch_bounds__(256) void flash_attn(
    const unsigned short* __restrict__ q,   // [B][H][T][D] bf16
    const unsigned short* __restrict__ k,
    const unsigned short* __restrict__ v,
    const float* __restrict__ bias01,       // [B][2][T]
    unsigned short* __restrict__ ctx) {     // [B][T][C] bf16
  __shared__ alignas(16) unsigned short sK[2][32][64];    // 8 KB
  __shared__ alignas(16) unsigned short sV[2][32][64];    // 8 KB
  __shared__ alignas(16) unsigned short sP[8][16][32];    // 8 KB (per-wave P relayout)
  __shared__ float sBias[32];

  const int tid = threadIdx.x, wave = tid >> 5, lane = tid & 31;
  const int l16 = lane & 15, hi = lane >> 4;
  const int bh = blockIdx.y, b = bh >> 3, h = bh & 7;
  const int qt0 = blockIdx.x * 128;
  const int qt  = qt0 + wave * 16;          // this wave's query rows
  const int cls = (h < 2) ? 0 : 1;
  const float* kbias = bias01 + ((size_t)b * 2 + cls) * TT;

  const unsigned short* qb = q + ((size_t)bh * TT) * DD;
  const unsigned short* kb = k + ((size_t)bh * TT) * DD;
  const unsigned short* vb = v + ((size_t)bh * TT) * DD;

  auto issueKV = [&](int kt, int buf) {
    tdm_load_2d_bf16(lds_off(&sK[buf][0][0]), kb + (size_t)kt * DD, 64, 32, 64, TT, 64);
    tdm_load_2d_bf16(lds_off(&sV[buf][0][0]), vb + (size_t)kt * DD, 64, 32, 64, TT, 64);
  };

  // Q A-fragments for D = 0..31 and 32..63 (kept in registers for whole loop)
  v16bf qa[2];
#pragma unroll
  for (int c = 0; c < 2; ++c) {
    const unsigned short* base = qb + (size_t)(qt + l16) * DD + c * 32;
    BF16Frag f;
    f.lo = *(const U4*)(base + hi * 8);
    f.hi = *(const U4*)(base + 16 + hi * 8);
    qa[c] = fragcast(f);
  }

  v8f o[4];
#pragma unroll
  for (int j = 0; j < 4; ++j) o[j] = vzero();
  float rmax[8], rsum[8];
#pragma unroll
  for (int r = 0; r < 8; ++r) { rmax[r] = -3e38f; rsum[r] = 0.f; }

  if (wave == 0) issueKV(0, 0);

  for (int kt = 0; kt < TT; kt += 32) {
    const int cur = (kt >> 5) & 1;
    if (wave == 0) {
      if (kt + 32 < TT) {
        issueKV(kt + 32, cur ^ 1);
        __builtin_amdgcn_s_wait_tensorcnt(2);   // current tile done; next in flight
      } else {
        __builtin_amdgcn_s_wait_tensorcnt(0);
      }
    }
    if (tid < 32) sBias[tid] = kbias[kt + tid];
    __syncthreads();

    // S = Q @ K^T for 16 queries x 32 keys (two 16-key n-tiles)
    v8f s[2];
#pragma unroll
    for (int jn = 0; jn < 2; ++jn) {
      int n = jn * 16 + l16;               // key row within tile
      v8f accv = vzero();
#pragma unroll
      for (int c = 0; c < 2; ++c) {        // chain over D chunks of 32
        BF16Frag f;
        f.lo = *(const U4*)&sK[cur][n][c * 32 + hi * 8];
        f.hi = *(const U4*)&sK[cur][n][c * 32 + 16 + hi * 8];
        accv = wmma_bf16(qa[c], fragcast(f), accv);
      }
      float bn = sBias[jn * 16 + l16];
#pragma unroll
      for (int r = 0; r < 8; ++r) accv[r] += bn;
      s[jn] = accv;
    }

    // online softmax per query row (rows live in 16-lane halves)
#pragma unroll
    for (int r = 0; r < 8; ++r) {
      float mloc = fmaxf(s[0][r], s[1][r]);
#pragma unroll
      for (int off = 1; off < 16; off <<= 1)
        mloc = fmaxf(mloc, __shfl_xor(mloc, off, 32));
      float mnew = fmaxf(rmax[r], mloc);
      float corr = exp2f((rmax[r] - mnew) * L2E);
      rmax[r] = mnew;
      float p0 = exp2f((s[0][r] - mnew) * L2E);
      float p1 = exp2f((s[1][r] - mnew) * L2E);
      float lsum = p0 + p1;
#pragma unroll
      for (int off = 1; off < 16; off <<= 1)
        lsum += __shfl_xor(lsum, off, 32);
      rsum[r] = rsum[r] * corr + lsum;
#pragma unroll
      for (int j = 0; j < 4; ++j) o[j][r] *= corr;
      s[0][r] = p0; s[1][r] = p1;
    }

    // P (C/D layout) -> per-wave LDS, row-major [16 q][32 k]
#pragma unroll
    for (int r = 0; r < 8; ++r) {
      sP[wave][r + hi * 8][l16]      = f2bf(s[0][r]);
      sP[wave][r + hi * 8][16 + l16] = f2bf(s[1][r]);
    }
    asm volatile("s_wait_dscnt 0" ::: "memory");  // wave-local LDS RAW

    // P as A-fragment (K = 32 keys)
    BF16Frag pf;
    pf.lo = *(const U4*)&sP[wave][l16][hi * 8];
    pf.hi = *(const U4*)&sP[wave][l16][16 + hi * 8];
    v16bf pa = fragcast(pf);

    // O += P @ V : four 16-wide D chunks; B-frag is transposed V
#pragma unroll
    for (int jd = 0; jd < 4; ++jd) {
      int d = jd * 16 + l16;
      US16 vbf;
#pragma unroll
      for (int vv = 0; vv < 8; ++vv) {
        int kk = 2 * vv + hi * 8 + (vv >= 4 ? 8 : 0);
        vbf.u[2 * vv]     = sV[cur][kk][d];
        vbf.u[2 * vv + 1] = sV[cur][kk + 1][d];
      }
      o[jd] = wmma_bf16(pa, fragcast16(vbf), o[jd]);
    }
    __syncthreads();   // WAR: tile kt+64 reuses this buffer
  }

  // normalize and store ctx[b][t][h*64+d] as bf16
#pragma unroll
  for (int jd = 0; jd < 4; ++jd) {
    int c = h * DD + jd * 16 + l16;
#pragma unroll
    for (int r = 0; r < 8; ++r) {
      int t = qt + r + hi * 8;
      float val = o[jd][r] * (1.0f / rsum[r]);
      ctx[((size_t)b * TT + t) * CC + c] = f2bf(val);
    }
  }
}

// ---------------------------------------------------------------------------
extern "C" void kernel_launch(void* const* d_in, const int* in_sizes, int n_in,
                              void* d_out, int out_size, void* d_ws, size_t ws_size,
                              hipStream_t stream) {
  (void)in_sizes; (void)n_in; (void)out_size; (void)ws_size;
  const float* x    = (const float*)d_in[0];
  const int*   mask = (const int*)d_in[1];
  const int*   cdrs = (const int*)d_in[2];
  const float* Wq   = (const float*)d_in[3];
  const float* bq   = (const float*)d_in[4];
  const float* Wk   = (const float*)d_in[5];
  const float* bk   = (const float*)d_in[6];
  const float* Wv   = (const float*)d_in[7];
  const float* bv   = (const float*)d_in[8];
  const float* Wo   = (const float*)d_in[9];
  const float* bo   = (const float*)d_in[10];
  float* out = (float*)d_out;

  char* ws = (char*)d_ws;
  size_t off = 0;
  auto alloc = [&](size_t bytes) -> char* {
    char* p = ws + off;
    off += (bytes + 255) & ~(size_t)255;
    return p;
  };
  unsigned short* x_bf = (unsigned short*)alloc((size_t)MM * CC * 2);  // 8 MB
  unsigned short* Wtq  = (unsigned short*)alloc((size_t)CC * CC * 2);
  unsigned short* Wtk  = (unsigned short*)alloc((size_t)CC * CC * 2);
  unsigned short* Wtv  = (unsigned short*)alloc((size_t)CC * CC * 2);
  unsigned short* Wto  = (unsigned short*)alloc((size_t)CC * CC * 2);
  unsigned short* qb   = (unsigned short*)alloc((size_t)MM * CC * 2);  // [B][H][T][D]
  unsigned short* kbuf = (unsigned short*)alloc((size_t)MM * CC * 2);
  unsigned short* vbuf = (unsigned short*)alloc((size_t)MM * CC * 2);
  unsigned short* ctx  = (unsigned short*)alloc((size_t)MM * CC * 2);  // [B][T][C]
  float* bias01        = (float*)alloc((size_t)BB * 2 * TT * 4);

  dim3 tB(16, 16), tG(CC / 16, CC / 16);
  transpose_cvt<<<tG, tB, 0, stream>>>(Wq, Wtq);
  transpose_cvt<<<tG, tB, 0, stream>>>(Wk, Wtk);
  transpose_cvt<<<tG, tB, 0, stream>>>(Wv, Wtv);
  transpose_cvt<<<tG, tB, 0, stream>>>(Wo, Wto);

  int nx = MM * CC;
  cvt_bf16<<<(nx + 255) / 256, 256, 0, stream>>>(x, x_bf, nx);
  prep_bias<<<BB, 256, 0, stream>>>(mask, cdrs, bias01);

  dim3 gG(CC / 64, MM / 128), gB(256);
  const float inv_sqrt_d = 0.125f;  // 1/sqrt(64), folded into Q
  gemm_wmma_bf16<<<gG, gB, 0, stream>>>(x_bf, Wtq, bq, qb,   nullptr, inv_sqrt_d, 0);
  gemm_wmma_bf16<<<gG, gB, 0, stream>>>(x_bf, Wtk, bk, kbuf, nullptr, 1.0f, 0);
  gemm_wmma_bf16<<<gG, gB, 0, stream>>>(x_bf, Wtv, bv, vbuf, nullptr, 1.0f, 0);

  flash_attn<<<dim3(TT / 128, BB * HH), 256, 0, stream>>>(qb, kbuf, vbuf, bias01, ctx);

  gemm_wmma_bf16<<<gG, gB, 0, stream>>>(ctx, Wto, bo, nullptr, out, 1.0f, 1);
}
